// SE_5686536699930
// MI455X (gfx1250) — matrix operand order repaired and checked by reading, not compiled, not run on hardware
//
#include <hip/hip_runtime.h>
#include <hip/hip_bf16.h>
#include <math.h>

// ---------------------------------------------------------------------------
// MI455X (gfx1250) GNN message passing, bf16 WMMA everywhere.
//   N=20000 nodes, D=16, E=320000 edges, HID=128.
// ~136 GFLOP vs ~0.8 GB of HBM traffic -> memory bound at 23.3 TB/s, provided
// the E x 384 concat and E x 128 message tensors are never materialized.
// Fused edge kernel: gather + edge1 + silu + edge2 + silu + segment_sum.
// Round-4 change: round-3's "all 18 loads first" sched_group_barrier caused
// accumulator spills to scratch (seen as Folded Reload/scratch_store in the
// disasm). Now: #pragma unroll 1 bounds each scheduling region to one K-step,
// and a gentle software pipeline (PRE loads, then (1 WMMA, 2 loads) x6, then
// 2 WMMAs) keeps only ~2 B fragments live -> graduated waits, no spills.
// ---------------------------------------------------------------------------

#define N_NODES 20000
#define DEG     16
#define N_EDGES (N_NODES * DEG)
#define HID     128

typedef __attribute__((ext_vector_type(16))) __bf16       v16bf;
typedef __attribute__((ext_vector_type(8)))  float        v8f;
typedef __attribute__((ext_vector_type(4)))  unsigned int v4u;

union FragBF { v16bf v; v4u q[2]; };

#if defined(__has_builtin)
#if __has_builtin(__builtin_amdgcn_sched_group_barrier)
#define SCHED_GROUP(mask, size) __builtin_amdgcn_sched_group_barrier(mask, size, 0)
#endif
#endif
#ifndef SCHED_GROUP
#define SCHED_GROUP(mask, size)
#endif
// masks: 0x008 = MFMA/WMMA, 0x020 = VMEM read, 0x100 = DS read

// Pipelined K-step shape: PRE loads, (1 WMMA, 2 loads) x6, 2 WMMAs.
// Load count = PRE + 12, WMMA count = 8.
template <int MASK, int PRE>
__device__ __forceinline__ void sched_pipe() {
  SCHED_GROUP(MASK, PRE);
  #pragma unroll
  for (int i = 0; i < 6; i++) { SCHED_GROUP(0x008, 1); SCHED_GROUP(MASK, 2); }
  SCHED_GROUP(0x008, 2);
}

__device__ __forceinline__ unsigned short f2bf(float f) {
  unsigned u = __float_as_uint(f);
  u += 0x7fffu + ((u >> 16) & 1u);          // round-to-nearest-even
  return (unsigned short)(u >> 16);
}
__device__ __forceinline__ float silu_f(float x) { return x / (1.0f + __expf(-x)); }

// CDNA5 async global->LDS copy (one b128 per lane), tracked by ASYNCcnt.
// VDST = per-lane LDS byte offset (generic LDS addr low 32 bits), VADDR = u64.
__device__ __forceinline__ void async_copy_b128(const void* g, unsigned lds_off) {
  asm volatile("global_load_async_to_lds_b128 %0, %1, off"
               :: "v"(lds_off), "v"((unsigned long long)(uintptr_t)g)
               : "memory");
}
#define WAIT_ASYNC_LE2() asm volatile("s_wait_asynccnt 0x2" ::: "memory")
#define WAIT_ASYNC_LE0() asm volatile("s_wait_asynccnt 0x0" ::: "memory")

// ---------------- small utility kernels ----------------
__global__ void k_zero(float* p, long n) {
  long i = (long)blockIdx.x * blockDim.x + threadIdx.x;
  if (i < n) p[i] = 0.0f;
}

__global__ void k_cast(const float* __restrict__ s, unsigned short* __restrict__ d, long n) {
  long i = (long)blockIdx.x * blockDim.x + threadIdx.x;
  if (i < n) d[i] = f2bf(s[i]);
}

// W: [K,128] f32 row-major  ->  Wt: [128,Kpad] bf16 (transposed, zero-padded)
__global__ void k_wt(const float* __restrict__ W, unsigned short* __restrict__ Wt,
                     int K, int Kpad) {
  int t = blockIdx.x * blockDim.x + threadIdx.x;
  if (t >= 128 * Kpad) return;
  int n = t / Kpad, k = t % Kpad;
  float v = (k < K) ? W[(long)k * 128 + n] : 0.0f;
  Wt[(long)n * Kpad + k] = f2bf(v);
}

// ---------------- edge geometric features ----------------
// A_ef[e][0..63] bf16: 0..15 radial RBF, 16..23 angular, 24..39 edge_attr,
// 40..63 zero pad (K padded to 64 for the WMMA K-loop).
__global__ void k_edgefeat(const float* __restrict__ pos,
                           const int* __restrict__ row, const int* __restrict__ col,
                           const float* __restrict__ eattr,
                           unsigned short* __restrict__ aef) {
  int e = blockIdx.x * blockDim.x + threadIdx.x;
  if (e >= N_EDGES) return;
  int i = row[e], j = col[e];
  float pjx = pos[3 * j], pjy = pos[3 * j + 1], pjz = pos[3 * j + 2];
  float rx = pos[3 * i] - pjx, ry = pos[3 * i + 1] - pjy, rz = pos[3 * i + 2] - pjz;
  float dist = sqrtf(rx * rx + ry * ry + rz * rz);

  unsigned short outv[40];
  const float rw = 8.0f / 15.0f;
  #pragma unroll
  for (int r = 0; r < 16; r++) {
    float c = dist - (float)r * rw;
    outv[r] = f2bf(__expf(-(c * c) / (rw * rw)));
  }
  float inv1 = 1.0f / (dist + 1e-8f);
  float v1x = rx * inv1, v1y = ry * inv1, v1z = rz * inv1;
  float ang[8];
  #pragma unroll
  for (int a = 0; a < 8; a++) ang[a] = 0.0f;
  float cnt = 0.0f;
  const float aw = 2.0f / 7.0f;
  for (int d = 0; d < DEG; d++) {
    int k = row[j * DEG + d];                 // nbr = row.reshape(N, D)
    float wx = pos[3 * k] - pjx, wy = pos[3 * k + 1] - pjy, wz = pos[3 * k + 2] - pjz;
    float inv2 = 1.0f / (sqrtf(wx * wx + wy * wy + wz * wz) + 1e-8f);
    float cosv = (v1x * wx + v1y * wy + v1z * wz) * inv2;
    if (k != i) {
      cnt += 1.0f;
      #pragma unroll
      for (int a = 0; a < 8; a++) {
        float c = cosv - (-1.0f + (float)a * aw);
        ang[a] += __expf(-(c * c) / (aw * aw));
      }
    }
  }
  float icnt = 1.0f / fmaxf(cnt, 1.0f);
  #pragma unroll
  for (int a = 0; a < 8; a++) outv[16 + a] = f2bf(ang[a] * icnt);
  #pragma unroll
  for (int q = 0; q < 16; q++) outv[24 + q] = f2bf(eattr[(long)e * 16 + q]);
  unsigned short* o = aef + (long)e * 64;
  #pragma unroll
  for (int q = 0; q < 40; q++) o[q] = outv[q];
  #pragma unroll
  for (int q = 40; q < 64; q++) o[q] = 0;
}

// ---------------- generic dense GEMM: [M,K]bf16 @ Wt[128,K] + bias ----------------
// One wave computes a 16x128 tile. K-loop kept rolled (#pragma unroll 1) so
// each scheduling region is exactly 18 VMEM reads + 8 WMMAs, pipelined via
// sched_pipe. flags: 1=silu, 2=write f32, 4=write bf16.
__global__ void k_gemm(const unsigned short* __restrict__ A,
                       const unsigned short* __restrict__ Wt,
                       const float* __restrict__ bias,
                       float* __restrict__ outf,
                       unsigned short* __restrict__ outb,
                       int M, int K, int flags) {
  int lane = threadIdx.x & 31, wv = threadIdx.x >> 5;
  int tile = blockIdx.x * (blockDim.x >> 5) + wv;
  if (tile * 16 >= M) return;
  int r = lane & 15, half = lane >> 4;
  const unsigned short* arow = A + (long)(tile * 16 + r) * K;

  v8f acc[8] = {};
  #pragma unroll 1
  for (int k0 = 0; k0 < K; k0 += 32) {
    FragBF a;                                   // A: contiguous k, one 16-elem run (as B)
    a.q[0] = *(const v4u*)(arow + k0 + 16 * half);
    a.q[1] = *(const v4u*)(arow + k0 + 16 * half + 8);
    FragBF b[8];
    #pragma unroll
    for (int nt = 0; nt < 8; nt++) {
      const unsigned short* bp = Wt + (long)(nt * 16 + r) * K + k0 + 16 * half;
      b[nt].q[0] = ((const v4u*)bp)[0];
      b[nt].q[1] = ((const v4u*)bp)[1];
    }
    #pragma unroll
    for (int nt = 0; nt < 8; nt++)
      acc[nt] = __builtin_amdgcn_wmma_f32_16x16x32_bf16(
          false, a.v, false, b[nt].v, (short)0, acc[nt], false, false);
    sched_pipe<0x020, 6>();                     // A(2)+b0,b1(4); then (W,2L)x6; W,W
  }
  #pragma unroll
  for (int nt = 0; nt < 8; nt++) {
    int n = nt * 16 + r;
    float bb = bias[n];
    #pragma unroll
    for (int rr = 0; rr < 8; rr++) {
      long m = (long)tile * 16 + rr + 8 * half;   // D layout: reg rr -> row rr + 8*half
      float val = acc[nt][rr] + bb;
      if (flags & 1) val = silu_f(val);
      if (flags & 2) outf[m * HID + n] = val;
      if (flags & 4) outb[m * HID + n] = f2bf(val);
    }
  }
}

// ---------------- fused edge MLP + segment_sum ----------------
// Per wave: 16 edges. GEMM1 A rows gathered on the fly from [h[row],h[col],ef].
// edge1 Wt staged per-32-K-step into LDS (8KB tiles), double-buffered with
// global_load_async_to_lds_b128 and shared by all 8 waves. GEMM1 B fragments
// come from ds_load_b128. GEMM1 -> silu -> LDS re-layout -> GEMM2 ->
// silu -> atomicAdd into agg[row] (10MB, L2-resident).
__global__ void k_edge_mlp(const unsigned short* __restrict__ hb,
                           const unsigned short* __restrict__ efb,
                           const int* __restrict__ row, const int* __restrict__ col,
                           const unsigned short* __restrict__ Wt1, const float* __restrict__ b1,
                           const unsigned short* __restrict__ Wt2, const float* __restrict__ b2,
                           float* __restrict__ agg) {
  __shared__ unsigned short wt1s[2 * 32 * 128];   // 2 x 8KB K-step tiles: [n]*32+[k]
  __shared__ unsigned short xch[8 * 16 * HID];    // 32KB D->A exchange
  int lane = threadIdx.x & 31, wv = threadIdx.x >> 5;
  int tile = blockIdx.x * 8 + wv;                 // E/16 tiles, exact
  int r = lane & 15, half = lane >> 4;
  int e_lane = tile * 16 + r;
  int re = row[e_lane], ce = col[e_lane];
  const unsigned short* seg0 = hb + (long)re * HID;
  const unsigned short* seg1 = hb + (long)ce * HID;
  const unsigned short* seg2 = efb + (long)e_lane * HID;

  // warm edge2 weights into cache while GEMM1 runs (global_prefetch_b8)
  __builtin_prefetch(Wt2 + (size_t)threadIdx.x * 64, 0, 1);
  __builtin_prefetch(Wt2 + (size_t)threadIdx.x * 64 + 32, 0, 1);

  // async-stage one 128x32 Wt1 tile (512 b128 chunks; 2 per thread)
  auto stage = [&](int k0, int buf) {
    #pragma unroll
    for (int it = 0; it < 2; it++) {
      int c = threadIdx.x + it * 256;
      int n = c >> 2, k8 = (c & 3) << 3;
      const unsigned short* src = Wt1 + (long)n * 384 + k0 + k8;
      unsigned dst = (unsigned)(uintptr_t)(wt1s + buf * (32 * 128) + n * 32 + k8);
      async_copy_b128(src, dst);
    }
  };

  stage(0, 0);                                    // prime the pipeline
  v8f acc[8] = {};
  #pragma unroll
  for (int ks = 0; ks < 12; ks++) {
    int k0 = ks * 32;
    int buf = ks & 1;
    // A fragment (gathered; 128-wide segments so 32-wide K-steps never straddle)
    const unsigned short* base = (k0 < 128) ? seg0 : (k0 < 256) ? seg1 : seg2;
    int koff = k0 & 127;
    FragBF a;
    a.q[0] = *(const v4u*)(base + koff + 16 * half);
    a.q[1] = *(const v4u*)(base + koff + 16 * half + 8);
    if (ks < 11) { stage(k0 + 32, buf ^ 1); WAIT_ASYNC_LE2(); }
    else         { WAIT_ASYNC_LE0(); }
    __syncthreads();                              // stage-ks tile visible to all waves
    const unsigned short* wb = wt1s + buf * (32 * 128);
    FragBF b[8];
    #pragma unroll
    for (int nt = 0; nt < 8; nt++) {
      const unsigned short* bp = wb + (nt * 16 + r) * 32 + 16 * half;
      b[nt].q[0] = ((const v4u*)bp)[0];
      b[nt].q[1] = ((const v4u*)bp)[1];
    }
    #pragma unroll
    for (int nt = 0; nt < 8; nt++)
      acc[nt] = __builtin_amdgcn_wmma_f32_16x16x32_bf16(
          false, a.v, false, b[nt].v, (short)0, acc[nt], false, false);
    sched_pipe<0x100, 4>();                       // b0,b1 DS; then (W,2L)x6; W,W
    __syncthreads();                              // done reading buf before re-stage
  }

  // bias + silu, D-layout -> row-major bf16 tile in LDS
  unsigned short* lt = xch + wv * 16 * HID;
  #pragma unroll
  for (int nt = 0; nt < 8; nt++) {
    int n = nt * 16 + r;
    float bb = b1[n];
    #pragma unroll
    for (int rr = 0; rr < 8; rr++) {
      int m = rr + 8 * half;
      lt[m * HID + n] = f2bf(silu_f(acc[nt][rr] + bb));
    }
  }
  __syncthreads();

  // GEMM2: (16x128) @ Wt2[128,128]; A from LDS (ds_load_b128), B global.
  // Rolled loop -> one K-step per scheduling region.
  const unsigned short* arow2 = lt + r * HID;
  v8f acc2[8] = {};
  #pragma unroll 1
  for (int k0 = 0; k0 < 128; k0 += 32) {
    FragBF a;
    a.q[0] = *(const v4u*)(arow2 + k0 + 16 * half);
    a.q[1] = *(const v4u*)(arow2 + k0 + 16 * half + 8);
    FragBF b[8];
    #pragma unroll
    for (int nt = 0; nt < 8; nt++) {
      const unsigned short* bp = Wt2 + (long)(nt * 16 + r) * 128 + k0 + 16 * half;
      b[nt].q[0] = ((const v4u*)bp)[0];
      b[nt].q[1] = ((const v4u*)bp)[1];
    }
    #pragma unroll
    for (int nt = 0; nt < 8; nt++)
      acc2[nt] = __builtin_amdgcn_wmma_f32_16x16x32_bf16(
          false, a.v, false, b[nt].v, (short)0, acc2[nt], false, false);
    SCHED_GROUP(0x100, 2);                        // A from LDS first
    sched_pipe<0x020, 4>();                       // then pipelined B + WMMAs
  }
  // bias + silu + scatter-add
  int tgt[8];
  #pragma unroll
  for (int rr = 0; rr < 8; rr++) tgt[rr] = row[tile * 16 + rr + 8 * half];
  #pragma unroll
  for (int nt = 0; nt < 8; nt++) {
    int n = nt * 16 + r;
    float bb = b2[n];
    #pragma unroll
    for (int rr = 0; rr < 8; rr++) {
      float val = silu_f(acc2[nt][rr] + bb);
      atomicAdd(agg + (long)tgt[rr] * HID + n, val);
    }
  }
}

// cat[n][0:128]=h_bf16, cat[n][128:256]=bf16(agg)
__global__ void k_concat(const unsigned short* __restrict__ hb,
                         const float* __restrict__ agg,
                         unsigned short* __restrict__ cat) {
  long t = (long)blockIdx.x * blockDim.x + threadIdx.x;
  if (t >= (long)N_NODES * HID) return;
  long nrow = t / HID, c = t % HID;
  cat[nrow * 256 + c] = hb[t];
  cat[nrow * 256 + 128 + c] = f2bf(agg[t]);
}

__global__ void k_colsum(const float* __restrict__ h, float* __restrict__ sumh) {
  int c = threadIdx.x & 127;
  int rpb = blockDim.x >> 7;
  int rstart = blockIdx.x * rpb + (threadIdx.x >> 7);
  int stride = gridDim.x * rpb;
  float s = 0.0f;
  for (int m = rstart; m < N_NODES; m += stride) s += h[(long)m * HID + c];
  atomicAdd(sumh + c, s);
}

__global__ void k_readout(const float* __restrict__ sumh, const float* __restrict__ W,
                          const float* __restrict__ b, float* __restrict__ out) {
  int c = threadIdx.x;
  if (c >= HID) return;
  float s = b[c];
  for (int k = 0; k < HID; k++) s += sumh[k] * W[k * HID + c];
  out[c] = s;
}

// ---------------------------------------------------------------------------
extern "C" void kernel_launch(void* const* d_in, const int* in_sizes, int n_in,
                              void* d_out, int out_size, void* d_ws, size_t ws_size,
                              hipStream_t stream) {
  (void)in_sizes; (void)n_in; (void)out_size; (void)ws_size;
  // inputs: x, pos, edge_index, edge_attr, then params flattened as a JAX
  // pytree (dicts in sorted-key order):
  //   4: edge_proj.b  5: edge_proj.w  6: global_readout.b  7: global_readout.w
  //   8+8l+{0..7}: layers[l] edge1.b, edge1.w, edge2.b, edge2.w,
  //                          node1.b, node1.w, node2.b, node2.w
  //   32: node_proj.b  33: node_proj.w
  const float* x     = (const float*)d_in[0];
  const float* pos   = (const float*)d_in[1];
  const int*   eidx  = (const int*)d_in[2];
  const float* eattr = (const float*)d_in[3];
  const int* row = eidx;
  const int* col = eidx + N_EDGES;
  const float* ep_b = (const float*)d_in[4];
  const float* ep_w = (const float*)d_in[5];
  const float* gr_b = (const float*)d_in[6];
  const float* gr_w = (const float*)d_in[7];
  const float* L[3][8];
  for (int l = 0; l < 3; l++)
    for (int q = 0; q < 8; q++) L[l][q] = (const float*)d_in[8 + l * 8 + q];
  const float* np_b = (const float*)d_in[32];
  const float* np_w = (const float*)d_in[33];

  // workspace carve (~157 MB)
  char* wsp = (char*)d_ws;
  size_t off = 0;
  auto alloc = [&](size_t bytes) -> void* {
    void* p = wsp + off;
    off += (bytes + 255) & ~(size_t)255;
    return p;
  };
  unsigned short* xb  = (unsigned short*)alloc((size_t)N_NODES * 64 * 2);
  unsigned short* hb  = (unsigned short*)alloc((size_t)N_NODES * HID * 2);
  unsigned short* aef = (unsigned short*)alloc((size_t)N_EDGES * 64 * 2);
  unsigned short* efb = (unsigned short*)alloc((size_t)N_EDGES * HID * 2);
  float*          agg = (float*)alloc((size_t)N_NODES * HID * 4);
  unsigned short* cat = (unsigned short*)alloc((size_t)N_NODES * 256 * 2);
  unsigned short* tb  = (unsigned short*)alloc((size_t)N_NODES * HID * 2);
  float*         sumh = (float*)alloc(HID * 4);
  unsigned short* wt_np = (unsigned short*)alloc(128 * 64 * 2);
  unsigned short* wt_ep = (unsigned short*)alloc(128 * 64 * 2);
  unsigned short *wt_e1[3], *wt_e2[3], *wt_n1[3], *wt_n2[3];
  for (int l = 0; l < 3; l++) {
    wt_e1[l] = (unsigned short*)alloc(128 * 384 * 2);
    wt_e2[l] = (unsigned short*)alloc(128 * 128 * 2);
    wt_n1[l] = (unsigned short*)alloc(128 * 256 * 2);
    wt_n2[l] = (unsigned short*)alloc(128 * 128 * 2);
  }

  float* out_h = (float*)d_out;                       // local_se [N,128]
  float* out_g = out_h + (size_t)N_NODES * HID;       // global_se [1,128]

  // ---- weight prep: transpose + bf16 cast (B fragments become contiguous) ----
  auto wt = [&](const float* W, unsigned short* Wt, int K, int Kpad) {
    int tot = 128 * Kpad;
    k_wt<<<(tot + 255) / 256, 256, 0, stream>>>(W, Wt, K, Kpad);
  };
  wt(np_w, wt_np, 64, 64);
  wt(ep_w, wt_ep, 40, 64);                            // 40 padded to 64
  for (int l = 0; l < 3; l++) {
    wt(L[l][1], wt_e1[l], 384, 384);
    wt(L[l][3], wt_e2[l], 128, 128);
    wt(L[l][5], wt_n1[l], 256, 256);
    wt(L[l][7], wt_n2[l], 128, 128);
  }
  {
    long n = (long)N_NODES * 64;
    k_cast<<<(int)((n + 255) / 256), 256, 0, stream>>>(x, xb, n);
  }

  auto gemm = [&](const unsigned short* A, const unsigned short* Wt2,
                  const float* bias, float* outf, unsigned short* outb,
                  int M, int K, int flags) {
    int tiles = M / 16;
    k_gemm<<<(tiles + 7) / 8, 256, 0, stream>>>(A, Wt2, bias, outf, outb, M, K, flags);
  };

  // h = node_proj(x)  (bf16 only; f32 h is never consumed until final layer)
  gemm(xb, wt_np, np_b, nullptr, hb, N_NODES, 64, 4);

  // geometric edge features + edge_proj
  k_edgefeat<<<(N_EDGES + 255) / 256, 256, 0, stream>>>(pos, row, col, eattr, aef);
  gemm(aef, wt_ep, ep_b, nullptr, efb, N_EDGES, 64, 4);

  // message passing layers
  for (int l = 0; l < 3; l++) {
    long n = (long)N_NODES * HID;
    k_zero<<<(int)((n + 255) / 256), 256, 0, stream>>>(agg, n);
    k_edge_mlp<<<N_EDGES / 128, 256, 0, stream>>>(hb, efb, row, col,
                                                  wt_e1[l], L[l][0],
                                                  wt_e2[l], L[l][2], agg);
    k_concat<<<(int)((n + 255) / 256), 256, 0, stream>>>(hb, agg, cat);
    gemm(cat, wt_n1[l], L[l][4], nullptr, tb, N_NODES, 256, 1 | 4);   // silu
    bool last = (l == 2);
    gemm(tb, wt_n2[l], L[l][6], last ? out_h : nullptr, hb, N_NODES, 128,
         (last ? 2 : 0) | 4);
  }

  // global readout: sum over nodes, then 128x128 matvec
  k_zero<<<1, 256, 0, stream>>>(sumh, (long)HID);
  k_colsum<<<128, 256, 0, stream>>>(out_h, sumh);
  k_readout<<<1, 128, 0, stream>>>(sumh, gr_w, gr_b, out_g);
}
